// TradeTransformer_69836168233266
// MI455X (gfx1250) — matrix operand order repaired and checked by reading, not compiled, not run on hardware
//
#include <hip/hip_runtime.h>
#include <hip/hip_bf16.h>
#include <math.h>

// ---------------------------------------------------------------------------
// Types for WMMA
// ---------------------------------------------------------------------------
typedef _Float16 v16h __attribute__((ext_vector_type(16)));
typedef float    v8f  __attribute__((ext_vector_type(8)));

#define TILE_M 64
#define TILE_N 64
#define TILE_K 32
#define LDS_PAD 48   // row stride (halfs): 96B rows keep b128-aligned fragment reads

// ---------------------------------------------------------------------------
// Generic batched WMMA GEMM:  C = alpha * A @ B^T  (+bias)(*rowscale)(+residual)(+=)
//   A: (M, Kd) f32 row-major, per-batch stride strideA
//   B: weights. transB==0: (N, Kd) row-major (torch Linear layout).
//      transB==1: (Kd, N) row-major (e.g. V in attention P@V).
//   out_mode 0: C[z*strideC + m*ldc + n]
//   out_mode 1: head-split   (m=(b,l), n=(h,d)) -> [b][h][l][d]   (batch==1)
//   out_mode 2: head-merge   (z=(b,h), m=l, n=d) -> [b][l][h*Dh+d]
// Pipelined: double-buffered LDS; next tile's global loads issue before the
// current tile's WMMAs. A sched_barrier pins all fragment ds_loads ahead of
// the WMMA chain so the four WMMAs wait on partial dscnt thresholds instead
// of full drains.
// ---------------------------------------------------------------------------
__global__ __launch_bounds__(128)
void k_gemm(const float* __restrict__ A, int lda, long long strideA,
            const float* __restrict__ Bw, int ldb, long long strideB, int transB,
            const float* __restrict__ bias,
            const float* __restrict__ residual,
            const float* __restrict__ rowscale,
            float alpha,
            float* __restrict__ C, int ldc, long long strideC,
            int M, int N, int Kd,
            int accum, int out_mode, int Ltok, int Hh, int Dh)
{
    __shared__ _Float16 At[2][TILE_M][LDS_PAD];
    __shared__ _Float16 Bt[2][TILE_N][LDS_PAD];

    const int z  = blockIdx.z;
    const float* Ag = A  + (size_t)z * strideA;
    const float* Bg = Bw + (size_t)z * strideB;
    const int m0 = blockIdx.y * TILE_M;
    const int n0 = blockIdx.x * TILE_N;
    const int tid  = threadIdx.x;
    const int lane = tid & 31;
    const int wv   = tid >> 5;
    const int lh   = lane & 15;
    const bool hi  = lane >= 16;

    // Uniform (scalar-branch) fast-path predicates: full tile in range, K tile
    // full, and 16B-aligned rows -> unguarded float4 (global_load_b128) loads.
    const bool kAl   = (Kd & (TILE_K - 1)) == 0;
    const bool fastA = kAl && (m0 + TILE_M <= M) && ((lda & 3) == 0);
    const bool fastB = kAl && (n0 + TILE_N <= N) && ((ldb & 3) == 0);

    float ra[16], rb[16];   // staging registers (reused across pipeline stages)

    auto loadA = [&](int kb) {
        const int tr = tid >> 1, tc = (tid & 1) * 16;
        if (fastA) {
            const float* src = Ag + (size_t)(m0 + tr) * lda + kb + tc;
#pragma unroll
            for (int i = 0; i < 4; i++) {
                float4 f = *(const float4*)(src + 4 * i);
                ra[4 * i + 0] = f.x; ra[4 * i + 1] = f.y;
                ra[4 * i + 2] = f.z; ra[4 * i + 3] = f.w;
            }
        } else {
            const int gm = m0 + tr;
#pragma unroll
            for (int i = 0; i < 16; i++) {
                const int k = kb + tc + i;
                ra[i] = (gm < M && k < Kd) ? Ag[(size_t)gm * lda + k] : 0.f;
            }
        }
    };
    auto loadB = [&](int kb) {
        if (!transB) {
            const int tr = tid >> 1, tc = (tid & 1) * 16;
            if (fastB) {
                const float* src = Bg + (size_t)(n0 + tr) * ldb + kb + tc;
#pragma unroll
                for (int i = 0; i < 4; i++) {
                    float4 f = *(const float4*)(src + 4 * i);
                    rb[4 * i + 0] = f.x; rb[4 * i + 1] = f.y;
                    rb[4 * i + 2] = f.z; rb[4 * i + 3] = f.w;
                }
            } else {
                const int gn = n0 + tr;
#pragma unroll
                for (int i = 0; i < 16; i++) {
                    const int k = kb + tc + i;
                    rb[i] = (gn < N && k < Kd) ? Bg[(size_t)gn * ldb + k] : 0.f;
                }
            }
        } else {
            const int tk = tid >> 2, tn = (tid & 3) * 16;
            if (fastB) {
                const float* src = Bg + (size_t)(kb + tk) * ldb + n0 + tn;
#pragma unroll
                for (int i = 0; i < 4; i++) {
                    float4 f = *(const float4*)(src + 4 * i);
                    rb[4 * i + 0] = f.x; rb[4 * i + 1] = f.y;
                    rb[4 * i + 2] = f.z; rb[4 * i + 3] = f.w;
                }
            } else {
                const int k = kb + tk;
#pragma unroll
                for (int i = 0; i < 16; i++) {
                    const int gn = n0 + tn + i;
                    rb[i] = (k < Kd && gn < N) ? Bg[(size_t)k * ldb + gn] : 0.f;
                }
            }
        }
    };
    auto storeA = [&](int buf) {
        const int tr = tid >> 1, tc = (tid & 1) * 16;
#pragma unroll
        for (int i = 0; i < 16; i++) At[buf][tr][tc + i] = (_Float16)ra[i];
    };
    auto storeB = [&](int buf) {
        if (!transB) {
            const int tr = tid >> 1, tc = (tid & 1) * 16;
#pragma unroll
            for (int i = 0; i < 16; i++) Bt[buf][tr][tc + i] = (_Float16)rb[i];
        } else {
            const int tk = tid >> 2, tn = (tid & 3) * 16;
#pragma unroll
            for (int i = 0; i < 16; i++) Bt[buf][tn + i][tk] = (_Float16)rb[i];
        }
    };

    v8f acc[4];
#pragma unroll
    for (int t = 0; t < 4; t++)
#pragma unroll
        for (int r = 0; r < 8; r++) acc[t][r] = 0.f;

    // prologue: stage first K tile
    loadA(0); loadB(0);
    storeA(0); storeB(0);
    __syncthreads();

    int cur = 0;
    for (int kb = 0; kb < Kd; kb += TILE_K) {
        const bool has_next = (kb + TILE_K) < Kd;
        if (has_next) { loadA(kb + TILE_K); loadB(kb + TILE_K); } // overlap w/ WMMA

        // ---- A fragment: 16x32, contiguous k = 16*h + s (matches B) ----
        v16h a;
        {
            const _Float16* ap = &At[cur][wv * 16 + lh][0];
            const int k0 = hi ? 16 : 0, k1 = hi ? 24 : 8;
#pragma unroll
            for (int i = 0; i < 8; i++) a[i]     = ap[k0 + i];
#pragma unroll
            for (int i = 0; i < 8; i++) a[8 + i] = ap[k1 + i];
        }
        // ---- all 4 B fragments up front, in distinct registers ----
        const _Float16* bp0 = &Bt[cur][ 0 + lh][hi ? 16 : 0];
        const _Float16* bp1 = &Bt[cur][16 + lh][hi ? 16 : 0];
        const _Float16* bp2 = &Bt[cur][32 + lh][hi ? 16 : 0];
        const _Float16* bp3 = &Bt[cur][48 + lh][hi ? 16 : 0];
        v16h b0, b1, b2, b3;
#pragma unroll
        for (int i = 0; i < 16; i++) b0[i] = bp0[i];
#pragma unroll
        for (int i = 0; i < 16; i++) b1[i] = bp1[i];
#pragma unroll
        for (int i = 0; i < 16; i++) b2[i] = bp2[i];
#pragma unroll
        for (int i = 0; i < 16; i++) b3[i] = bp3[i];

        // Hard scheduling fence: keep all fragment ds_loads issued before the
        // WMMA chain so waits become partial dscnt thresholds, not full drains.
        __builtin_amdgcn_sched_barrier(0);

        // ---- 4 WMMAs back-to-back on the XDL pipe ----
        acc[0] = __builtin_amdgcn_wmma_f32_16x16x32_f16(
            false, a, false, b0, (short)0, acc[0], false, false);
        acc[1] = __builtin_amdgcn_wmma_f32_16x16x32_f16(
            false, a, false, b1, (short)0, acc[1], false, false);
        acc[2] = __builtin_amdgcn_wmma_f32_16x16x32_f16(
            false, a, false, b2, (short)0, acc[2], false, false);
        acc[3] = __builtin_amdgcn_wmma_f32_16x16x32_f16(
            false, a, false, b3, (short)0, acc[3], false, false);

        if (has_next) {
            storeA(cur ^ 1); storeB(cur ^ 1);
            __syncthreads();
            cur ^= 1;
        }
    }

    // ---- epilogue ----
#pragma unroll
    for (int nt = 0; nt < 4; nt++) {
        int gn = n0 + nt * 16 + lh;
        if (gn >= N) continue;
#pragma unroll
        for (int r = 0; r < 8; r++) {
            int gm = m0 + wv * 16 + (hi ? 8 : 0) + r;
            if (gm >= M) continue;
            float val = acc[nt][r] * alpha;
            if (bias)     val += bias[gn];
            if (rowscale) val *= rowscale[gm];
            size_t idx;
            if (out_mode == 1) {
                int b_ = gm / Ltok, l_ = gm % Ltok;
                int h_ = gn / Dh,   d_ = gn % Dh;
                idx = (((size_t)b_ * Hh + h_) * Ltok + l_) * Dh + d_;
            } else if (out_mode == 2) {
                int b_ = z / Hh, h_ = z % Hh;
                idx = ((size_t)b_ * Ltok + gm) * (size_t)(Hh * Dh)
                    + (size_t)h_ * Dh + gn;
            } else {
                idx = (size_t)z * strideC + (size_t)gm * ldc + gn;
            }
            if (residual) val += residual[idx];
            if (accum) C[idx] += val; else C[idx] = val;
        }
    }
}

// ---------------------------------------------------------------------------
// RMSNorm: one block per row
// ---------------------------------------------------------------------------
__global__ void k_rms(const float* __restrict__ x, const float* __restrict__ w,
                      float* __restrict__ y, int D)
{
    __shared__ float red[256];
    const int row = blockIdx.x;
    const float* xr = x + (size_t)row * D;
    float* yr = y + (size_t)row * D;
    float ss = 0.f;
    for (int i = threadIdx.x; i < D; i += blockDim.x) { float v = xr[i]; ss += v * v; }
    red[threadIdx.x] = ss; __syncthreads();
    for (int s = blockDim.x >> 1; s > 0; s >>= 1) {
        if (threadIdx.x < (unsigned)s) red[threadIdx.x] += red[threadIdx.x + s];
        __syncthreads();
    }
    const float scale = rsqrtf(red[0] / (float)D + 1.1920929e-07f);
    for (int i = threadIdx.x; i < D; i += blockDim.x) yr[i] = xr[i] * scale * w[i];
}

// ---------------------------------------------------------------------------
// RoPE (reference bug faithfully kept: both slots get cos):
//   dst[d]    = add[d]    + (x[d] - x[d+32]) * cos * scale
//   dst[d+32] = add[d+32] + (x[d+32] + x[d]) * cos * scale
// One thread per rotation pair; safe for dst==src.
// ---------------------------------------------------------------------------
__global__ void k_rope(const float* __restrict__ src, const float* __restrict__ addv,
                       float* __restrict__ dst, const float* __restrict__ ts,
                       int Bb, int Hh, int Ll, float scale)
{
    int idx = blockIdx.x * blockDim.x + threadIdx.x;
    int total = Bb * Hh * Ll * 32;
    if (idx >= total) return;
    int d = idx & 31;
    int l = (idx >> 5) % Ll;
    int h = ((idx >> 5) / Ll) % Hh;
    int b = (idx >> 5) / (Ll * Hh);
    size_t base = (((size_t)b * Hh + h) * Ll + l) * 64;
    float t = ts[(size_t)b * Ll + l];
    float freq = powf(10000.0f, -((float)(2 * d)) * (1.0f / 64.0f));
    float c = cosf(t * freq);
    float x1 = src[base + d], x2 = src[base + d + 32];
    float r1 = (x1 - x2) * c * scale;
    float r2 = (x2 + x1) * c * scale;
    float a1 = addv ? addv[base + d]      : 0.f;
    float a2 = addv ? addv[base + d + 32] : 0.f;
    dst[base + d]      = a1 + r1;
    dst[base + d + 32] = a2 + r2;
}

// ---------------------------------------------------------------------------
// Row softmax (optional causal mask: col > (row % Mq) gets -1e9)
// ---------------------------------------------------------------------------
__global__ void k_softmax(float* __restrict__ S, int Nk, int Mq, int causal)
{
    __shared__ float red[256];
    const int row = blockIdx.x;
    const int q = row % Mq;
    float* p = S + (size_t)row * Nk;
    float mx = -3.4e38f;
    for (int i = threadIdx.x; i < Nk; i += blockDim.x) {
        float v = p[i]; if (causal && i > q) v -= 1e9f;
        mx = fmaxf(mx, v);
    }
    red[threadIdx.x] = mx; __syncthreads();
    for (int s = blockDim.x >> 1; s > 0; s >>= 1) {
        if (threadIdx.x < (unsigned)s)
            red[threadIdx.x] = fmaxf(red[threadIdx.x], red[threadIdx.x + s]);
        __syncthreads();
    }
    mx = red[0]; __syncthreads();
    float sum = 0.f;
    for (int i = threadIdx.x; i < Nk; i += blockDim.x) {
        float v = p[i]; if (causal && i > q) v -= 1e9f;
        float e = __expf(v - mx); p[i] = e; sum += e;
    }
    red[threadIdx.x] = sum; __syncthreads();
    for (int s = blockDim.x >> 1; s > 0; s >>= 1) {
        if (threadIdx.x < (unsigned)s) red[threadIdx.x] += red[threadIdx.x + s];
        __syncthreads();
    }
    const float inv = 1.f / red[0];
    for (int i = threadIdx.x; i < Nk; i += blockDim.x) p[i] *= inv;
}

// ---------------------------------------------------------------------------
// MoE gate: softmax over 8 logits, top-2 (first occurrence), renormalize.
// we layout: [expert][row]  (row-scale vectors for the expert GEMM epilogue)
// ---------------------------------------------------------------------------
__global__ void k_gate(const float* __restrict__ logits, int ldl, int M,
                       float* __restrict__ we)
{
    int m = blockIdx.x * blockDim.x + threadIdx.x;
    if (m >= M) return;
    float l[8]; float mx = -3.4e38f;
    for (int e = 0; e < 8; e++) { l[e] = logits[(size_t)m * ldl + e]; mx = fmaxf(mx, l[e]); }
    float s = 0.f;
    for (int e = 0; e < 8; e++) { l[e] = __expf(l[e] - mx); s += l[e]; }
    for (int e = 0; e < 8; e++) l[e] /= s;
    int i0 = 0;
    for (int e = 1; e < 8; e++) if (l[e] > l[i0]) i0 = e;
    int i1 = (i0 == 0) ? 1 : 0;
    for (int e = 0; e < 8; e++) if (e != i0 && l[e] > l[i1]) i1 = e;
    float t = l[i0] + l[i1];
    for (int e = 0; e < 8; e++)
        we[(size_t)e * M + m] = (e == i0) ? l[i0] / t : ((e == i1) ? l[i1] / t : 0.f);
}

// ---------------------------------------------------------------------------
// SwiGLU: g[m][j] = silu(h[m][j]) * h[m][dff + j]
// ---------------------------------------------------------------------------
__global__ void k_swiglu(const float* __restrict__ h, float* __restrict__ g,
                         long long total, int dff)
{
    long long idx = (long long)blockIdx.x * blockDim.x + threadIdx.x;
    if (idx >= total) return;
    long long m = idx / dff; int j = (int)(idx % dff);
    float a = h[m * 2 * dff + j];
    float b = h[m * 2 * dff + dff + j];
    g[idx] = (a / (1.f + __expf(-a))) * b;
}

// ---------------------------------------------------------------------------
// Decoder input: d[b,0] = bos + te ; d[b,k>0] = prev[b,k-1] + te
// ---------------------------------------------------------------------------
__global__ void k_dec_init(const float* __restrict__ prev, const float* __restrict__ bos,
                           const float* __restrict__ time_w, const float* __restrict__ time_b,
                           const float* __restrict__ dts, float* __restrict__ dx,
                           int Bb, int Kk, int D)
{
    int idx = blockIdx.x * blockDim.x + threadIdx.x;
    int total = Bb * Kk * D;
    if (idx >= total) return;
    int n = idx % D;
    int k = (idx / D) % Kk;
    int b = idx / (D * Kk);
    float t = dts[b * Kk + k];
    float base = (k == 0) ? bos[n] : prev[((size_t)b * (Kk - 1) + (k - 1)) * D + n];
    dx[idx] = base + t * time_w[n] + time_b[n];
}

// ---------------------------------------------------------------------------
// Host-side launcher helpers
// ---------------------------------------------------------------------------
static inline void gemm(hipStream_t st,
                        const float* A, int lda, long long sA,
                        const float* Bw, int ldb, long long sB, int transB,
                        const float* bias, const float* res, const float* rsc, float alpha,
                        float* C, int ldc, long long sC,
                        int M, int N, int Kd, int batch,
                        int accum, int mode, int Ltok, int Hh, int Dh)
{
    dim3 g((N + TILE_N - 1) / TILE_N, (M + TILE_M - 1) / TILE_M, batch);
    k_gemm<<<g, 128, 0, st>>>(A, lda, sA, Bw, ldb, sB, transB,
                              bias, res, rsc, alpha, C, ldc, sC,
                              M, N, Kd, accum, mode, Ltok, Hh, Dh);
}

extern "C" void kernel_launch(void* const* d_in, const int* in_sizes, int n_in,
                              void* d_out, int out_size, void* d_ws, size_t ws_size,
                              hipStream_t stream)
{
    (void)in_sizes; (void)n_in; (void)out_size; (void)ws_size;
    const int Bx = 4, L = 1024, Kq = 5, DIN = 256, DM = 512, H = 8, DC = 128;
    const int DFF = 1024, NEXP = 8, HD = 64, VOCAB = 500;
    const int NT = Bx * L;     // 4096 encoder tokens
    const int ND = Bx * Kq;    // 20 decoder tokens
    const float SCALE = 0.125f; // 64^-0.5

    auto ip = [&](int i) { return (const float*)d_in[i]; };
    const float* market  = ip(0);
    const float* enc_ts  = ip(1);
    const float* dec_ts  = ip(2);
    const float* teach   = ip(3);
    const float* in_w    = ip(4);
    const float* in_b    = ip(5);
    const float* teach_w = ip(6);
    const float* time_w  = ip(7);
    const float* time_b  = ip(8);
    const float* bosv    = ip(9);
    const float* out_w   = ip(10);
    const float* out_b   = ip(11);
    const int ENC0 = 12;            // 6 layers x 11 leaves
    const int DEC0 = 12 + 6 * 11;   // 6 layers x 16 leaves

    // ---- workspace carve-out ----
    char* wsb = (char*)d_ws;
    size_t off = 0;
    auto alloc = [&](size_t elems) -> float* {
        float* p = (float*)(wsb + off);
        off += ((elems * sizeof(float) + 255) & ~(size_t)255);
        return p;
    };
    float* s_x   = alloc((size_t)NT * DM);        // residual stream / enc_out
    float* s_xn  = alloc((size_t)NT * DM);
    float* s_c   = alloc((size_t)NT * DC);
    float* s_q   = alloc((size_t)NT * DM);        // (B,H,L,64)
    float* s_kc  = alloc((size_t)NT * DM);
    float* s_ks  = alloc((size_t)NT * DM);        // k_r then ksum = k_c + rope(k_r)
    float* s_v   = alloc((size_t)NT * DM);
    float* s_att = alloc((size_t)NT * DM);
    float* s_h   = alloc((size_t)NT * 2 * DFF);
    float* s_g   = alloc((size_t)NT * DFF);
    float* s_gl  = alloc((size_t)NT * NEXP);
    float* s_we  = alloc((size_t)NEXP * NT);
    float* s_sc  = alloc((size_t)Bx * H * L * L); // attention scores (134 MB)
    float* s_prev = alloc((size_t)Bx * (Kq - 1) * DM);
    float* s_dx  = alloc((size_t)ND * DM);
    float* s_dxn = alloc((size_t)ND * DM);
    float* s_dc  = alloc((size_t)ND * DC);
    float* s_dq  = alloc((size_t)ND * DM);
    float* s_dkc = alloc((size_t)ND * DM);
    float* s_dks = alloc((size_t)ND * DM);
    float* s_dv  = alloc((size_t)ND * DM);
    float* s_dat = alloc((size_t)ND * DM);

    // ================= encoder =================
    // x = market @ in_w^T + in_b
    gemm(stream, market, DIN, 0, in_w, DIN, 0, 0, in_b, nullptr, nullptr, 1.f,
         s_x, DM, 0, NT, DM, DIN, 1, 0, 0, 0, 0, 0);

    for (int ly = 0; ly < 6; ly++) {
        int pb = ENC0 + ly * 11;
        const float* attn_norm = ip(pb + 0);
        const float* w_kv_c    = ip(pb + 1);
        const float* w_kc_up   = ip(pb + 2);
        const float* w_vc_up   = ip(pb + 3);
        const float* w_qr      = ip(pb + 4);
        const float* w_kr      = ip(pb + 5);
        const float* w_o       = ip(pb + 6);
        const float* ffn_norm  = ip(pb + 7);
        const float* gatew     = ip(pb + 8);
        const float* moe_w1    = ip(pb + 9);
        const float* moe_w2    = ip(pb + 10);

        k_rms<<<NT, 256, 0, stream>>>(s_x, attn_norm, s_xn, DM);
        // latent + projections (head-split written directly)
        gemm(stream, s_xn, DM, 0, w_kv_c, DM, 0, 0, nullptr, nullptr, nullptr, 1.f,
             s_c, DC, 0, NT, DC, DM, 1, 0, 0, 0, 0, 0);
        gemm(stream, s_c, DC, 0, w_kc_up, DC, 0, 0, nullptr, nullptr, nullptr, 1.f,
             s_kc, 0, 0, NT, DM, DC, 1, 0, 1, L, H, HD);
        gemm(stream, s_c, DC, 0, w_vc_up, DC, 0, 0, nullptr, nullptr, nullptr, 1.f,
             s_v, 0, 0, NT, DM, DC, 1, 0, 1, L, H, HD);
        gemm(stream, s_xn, DM, 0, w_qr, DM, 0, 0, nullptr, nullptr, nullptr, 1.f,
             s_q, 0, 0, NT, DM, DM, 1, 0, 1, L, H, HD);
        gemm(stream, s_xn, DM, 0, w_kr, DM, 0, 0, nullptr, nullptr, nullptr, 1.f,
             s_ks, 0, 0, NT, DM, DM, 1, 0, 1, L, H, HD);
        // rope: q (scaled), ksum = k_c + rope(k_r)
        int rt = Bx * H * L * 32;
        k_rope<<<(rt + 255) / 256, 256, 0, stream>>>(s_q, nullptr, s_q, enc_ts, Bx, H, L, SCALE);
        k_rope<<<(rt + 255) / 256, 256, 0, stream>>>(s_ks, s_kc, s_ks, enc_ts, Bx, H, L, 1.f);
        // scores = q @ ksum^T  (batched over B*H)
        gemm(stream, s_q, HD, (long long)L * HD, s_ks, HD, (long long)L * HD, 0,
             nullptr, nullptr, nullptr, 1.f, s_sc, L, (long long)L * L,
             L, L, HD, Bx * H, 0, 0, 0, 0, 0);
        k_softmax<<<Bx * H * L, 256, 0, stream>>>(s_sc, L, L, 0);
        // out = P @ V  (B stored (K,N): transB), head-merge into (B,L,512)
        gemm(stream, s_sc, L, (long long)L * L, s_v, HD, (long long)L * HD, 1,
             nullptr, nullptr, nullptr, 1.f, s_att, 0, 0,
             L, HD, L, Bx * H, 0, 2, L, H, HD);
        // x = x + attn @ w_o^T
        gemm(stream, s_att, DM, 0, w_o, DM, 0, 0, nullptr, s_x, nullptr, 1.f,
             s_x, DM, 0, NT, DM, DM, 1, 0, 0, 0, 0, 0);
        // MoE
        k_rms<<<NT, 256, 0, stream>>>(s_x, ffn_norm, s_xn, DM);
        gemm(stream, s_xn, DM, 0, gatew, DM, 0, 0, nullptr, nullptr, nullptr, 1.f,
             s_gl, NEXP, 0, NT, NEXP, DM, 1, 0, 0, 0, 0, 0);
        k_gate<<<(NT + 255) / 256, 256, 0, stream>>>(s_gl, NEXP, NT, s_we);
        for (int e = 0; e < NEXP; e++) {
            gemm(stream, s_xn, DM, 0, moe_w1 + (size_t)e * 2 * DFF * DM, DM, 0, 0,
                 nullptr, nullptr, nullptr, 1.f, s_h, 2 * DFF, 0,
                 NT, 2 * DFF, DM, 1, 0, 0, 0, 0, 0);
            long long tot = (long long)NT * DFF;
            k_swiglu<<<(int)((tot + 255) / 256), 256, 0, stream>>>(s_h, s_g, tot, DFF);
            gemm(stream, s_g, DFF, 0, moe_w2 + (size_t)e * DM * DFF, DFF, 0, 0,
                 nullptr, nullptr, s_we + (size_t)e * NT, 1.f,
                 s_x, DM, 0, NT, DM, DFF, 1, 1, 0, 0, 0, 0);
        }
    }
    // s_x now holds enc_out

    // ================= decoder inputs =================
    // prev = teacher[:, :K-1] @ teach_w^T   (batched over B, rows 0..3 of each (K,V) slab)
    gemm(stream, teach, VOCAB, (long long)Kq * VOCAB, teach_w, VOCAB, 0, 0,
         nullptr, nullptr, nullptr, 1.f, s_prev, DM, (long long)(Kq - 1) * DM,
         Kq - 1, DM, VOCAB, Bx, 0, 0, 0, 0, 0);
    {
        int tot = ND * DM;
        k_dec_init<<<(tot + 255) / 256, 256, 0, stream>>>(
            s_prev, bosv, time_w, time_b, dec_ts, s_dx, Bx, Kq, DM);
    }

    // ================= decoder =================
    for (int ly = 0; ly < 6; ly++) {
        int pb = DEC0 + ly * 16;
        const float* attn_norm  = ip(pb + 0);
        const float* w_kv_c     = ip(pb + 1);
        const float* w_kc_up    = ip(pb + 2);
        const float* w_vc_up    = ip(pb + 3);
        const float* w_qr       = ip(pb + 4);
        const float* w_kr       = ip(pb + 5);
        const float* w_o        = ip(pb + 6);
        const float* ffn_norm   = ip(pb + 7);
        const float* gatew      = ip(pb + 8);
        const float* moe_w1     = ip(pb + 9);
        const float* moe_w2     = ip(pb + 10);
        const float* cross_norm = ip(pb + 11);
        const float* cin_w      = ip(pb + 12);
        const float* cin_b      = ip(pb + 13);
        const float* out_w_c    = ip(pb + 14);
        const float* out_b_c    = ip(pb + 15);

        // ---- MLA self-attention (causal, 20 tokens) ----
        k_rms<<<ND, 256, 0, stream>>>(s_dx, attn_norm, s_dxn, DM);
        gemm(stream, s_dxn, DM, 0, w_kv_c, DM, 0, 0, nullptr, nullptr, nullptr, 1.f,
             s_dc, DC, 0, ND, DC, DM, 1, 0, 0, 0, 0, 0);
        gemm(stream, s_dc, DC, 0, w_kc_up, DC, 0, 0, nullptr, nullptr, nullptr, 1.f,
             s_dkc, 0, 0, ND, DM, DC, 1, 0, 1, Kq, H, HD);
        gemm(stream, s_dc, DC, 0, w_vc_up, DC, 0, 0, nullptr, nullptr, nullptr, 1.f,
             s_dv, 0, 0, ND, DM, DC, 1, 0, 1, Kq, H, HD);
        gemm(stream, s_dxn, DM, 0, w_qr, DM, 0, 0, nullptr, nullptr, nullptr, 1.f,
             s_dq, 0, 0, ND, DM, DM, 1, 0, 1, Kq, H, HD);
        gemm(stream, s_dxn, DM, 0, w_kr, DM, 0, 0, nullptr, nullptr, nullptr, 1.f,
             s_dks, 0, 0, ND, DM, DM, 1, 0, 1, Kq, H, HD);
        int rt = Bx * H * Kq * 32;
        k_rope<<<(rt + 255) / 256, 256, 0, stream>>>(s_dq, nullptr, s_dq, dec_ts, Bx, H, Kq, SCALE);
        k_rope<<<(rt + 255) / 256, 256, 0, stream>>>(s_dks, s_dkc, s_dks, dec_ts, Bx, H, Kq, 1.f);
        gemm(stream, s_dq, HD, (long long)Kq * HD, s_dks, HD, (long long)Kq * HD, 0,
             nullptr, nullptr, nullptr, 1.f, s_sc, Kq, (long long)Kq * Kq,
             Kq, Kq, HD, Bx * H, 0, 0, 0, 0, 0);
        k_softmax<<<Bx * H * Kq, 256, 0, stream>>>(s_sc, Kq, Kq, 1);
        gemm(stream, s_sc, Kq, (long long)Kq * Kq, s_dv, HD, (long long)Kq * HD, 1,
             nullptr, nullptr, nullptr, 1.f, s_dat, 0, 0,
             Kq, HD, Kq, Bx * H, 0, 2, Kq, H, HD);
        gemm(stream, s_dat, DM, 0, w_o, DM, 0, 0, nullptr, s_dx, nullptr, 1.f,
             s_dx, DM, 0, ND, DM, DM, 1, 0, 0, 0, 0, 0);

        // ---- cross-attention over enc_out ----
        k_rms<<<ND, 256, 0, stream>>>(s_dx, cross_norm, s_dxn, DM);
        gemm(stream, s_dxn, DM, 0, cin_w, DM, 0, 0, cin_b, nullptr, nullptr, 1.f,
             s_dq, 0, 0, ND, DM, DM, 1, 0, 1, Kq, H, HD);
        gemm(stream, s_x, DM, 0, cin_w + (size_t)DM * DM, DM, 0, 0, cin_b + DM,
             nullptr, nullptr, 1.f, s_kc, 0, 0, NT, DM, DM, 1, 0, 1, L, H, HD);
        gemm(stream, s_x, DM, 0, cin_w + (size_t)2 * DM * DM, DM, 0, 0, cin_b + 2 * DM,
             nullptr, nullptr, 1.f, s_v, 0, 0, NT, DM, DM, 1, 0, 1, L, H, HD);
        // scores = SCALE * q @ k^T
        gemm(stream, s_dq, HD, (long long)Kq * HD, s_kc, HD, (long long)L * HD, 0,
             nullptr, nullptr, nullptr, SCALE, s_sc, L, (long long)Kq * L,
             Kq, L, HD, Bx * H, 0, 0, 0, 0, 0);
        k_softmax<<<Bx * H * Kq, 256, 0, stream>>>(s_sc, L, Kq, 0);
        gemm(stream, s_sc, L, (long long)Kq * L, s_v, HD, (long long)L * HD, 1,
             nullptr, nullptr, nullptr, 1.f, s_dat, 0, 0,
             Kq, HD, L, Bx * H, 0, 2, Kq, H, HD);
        gemm(stream, s_dat, DM, 0, out_w_c, DM, 0, 0, out_b_c, s_dx, nullptr, 1.f,
             s_dx, DM, 0, ND, DM, DM, 1, 0, 0, 0, 0, 0);

        // ---- MoE (20 tokens) ----
        k_rms<<<ND, 256, 0, stream>>>(s_dx, ffn_norm, s_dxn, DM);
        gemm(stream, s_dxn, DM, 0, gatew, DM, 0, 0, nullptr, nullptr, nullptr, 1.f,
             s_gl, NEXP, 0, ND, NEXP, DM, 1, 0, 0, 0, 0, 0);
        k_gate<<<1, 256, 0, stream>>>(s_gl, NEXP, ND, s_we);
        for (int e = 0; e < NEXP; e++) {
            gemm(stream, s_dxn, DM, 0, moe_w1 + (size_t)e * 2 * DFF * DM, DM, 0, 0,
                 nullptr, nullptr, nullptr, 1.f, s_h, 2 * DFF, 0,
                 ND, 2 * DFF, DM, 1, 0, 0, 0, 0, 0);
            long long tot = (long long)ND * DFF;
            k_swiglu<<<(int)((tot + 255) / 256), 256, 0, stream>>>(s_h, s_g, tot, DFF);
            gemm(stream, s_g, DFF, 0, moe_w2 + (size_t)e * DM * DFF, DFF, 0, 0,
                 nullptr, nullptr, s_we + (size_t)e * ND, 1.f,
                 s_dx, DM, 0, ND, DM, DFF, 1, 1, 0, 0, 0, 0);
        }
    }

    // ================= output head =================
    gemm(stream, s_dx, DM, 0, out_w, DM, 0, 0, out_b, nullptr, nullptr, 1.f,
         (float*)d_out, VOCAB, 0, ND, VOCAB, DM, 1, 0, 0, 0, 0, 0);
}
